// NewNet_27814208209005
// MI455X (gfx1250) — matrix operand order repaired and checked by reading, not compiled, not run on hardware
//
#include <hip/hip_runtime.h>
#include <hip/hip_bf16.h>
#include <stddef.h>

// ---------------------------------------------------------------------------
// 2-layer LSTM, T=512, B=128, I=H=512.  Persistent-kernel design:
//   - 64 blocks x 256 threads (8 waves of 32).  Blocks 0..31 = layer 0,
//     blocks 32..63 = layer 1, software-pipelined one step apart.
//   - Each wave owns one 16x16 (batch x h) cell tile and computes the four
//     gate tiles (i,f,g,o) with v_wmma_f32_16x16x32_f16, K = I + H = 1024.
//   - A operand (x[t] ++ h[t-1], f16) staged in LDS per block; B operand
//     (packed f16 weights, [layer][gate-col][k] contiguous) streamed from L2.
//   - Grid-wide step barrier via atomics in workspace (graph-capture safe).
// ---------------------------------------------------------------------------

typedef __attribute__((ext_vector_type(16))) _Float16 hvec16;
typedef __attribute__((ext_vector_type(8)))  _Float16 hvec8;
typedef __attribute__((ext_vector_type(8)))  float    fvec8;

#define Tn   512
#define Bn   128
#define Hn   512
#define Kn   1024            // I + H
#define G4H  2048            // 4*H
#define NBLK 64
#define AROW 1032            // LDS row stride in f16 (1024 + 8 pad -> kills bank conflicts)

// ---------------- prep: pack weights to f16, K contiguous per gate column ----
__global__ void pack_weights(const float* __restrict__ Wih0, const float* __restrict__ Whh0,
                             const float* __restrict__ bih0, const float* __restrict__ bhh0,
                             const float* __restrict__ Wih1, const float* __restrict__ Whh1,
                             const float* __restrict__ bih1, const float* __restrict__ bhh1,
                             _Float16* __restrict__ wpack, float* __restrict__ bsum)
{
    unsigned idx = blockIdx.x * 256u + threadIdx.x;       // [l][g][k]
    if (idx >= 2u * G4H * Kn) return;
    unsigned l = idx >> 21;
    unsigned r = idx & ((1u << 21) - 1u);
    unsigned g = r >> 10;
    unsigned k = r & 1023u;
    const float* Wih = l ? Wih1 : Wih0;
    const float* Whh = l ? Whh1 : Whh0;
    float v = (k < 512u) ? Wih[(size_t)g * 512 + k] : Whh[(size_t)g * 512 + (k - 512u)];
    wpack[idx] = (_Float16)v;
    if (k == 0u) {
        const float* bi = l ? bih1 : bih0;
        const float* bh = l ? bhh1 : bhh0;
        bsum[l * G4H + g] = bi[g] + bh[g];
    }
}

__global__ void zero_u32(unsigned* __restrict__ p, size_t n)
{
    size_t i = (size_t)blockIdx.x * 256 + threadIdx.x;
    if (i < n) p[i] = 0u;
}

__device__ __forceinline__ float fast_sigmoid(float x)
{
    return 1.0f / (1.0f + __expf(-x));
}
__device__ __forceinline__ float fast_tanh(float x)
{
    float e = __expf(-2.0f * x);
    return (1.0f - e) / (1.0f + e);
}

// ---------------- persistent fused 2-layer LSTM --------------------------
__global__ __launch_bounds__(256) void lstm_persistent(
    const float*    __restrict__ x,       // (T,B,I) f32
    const _Float16* __restrict__ wpack,   // [2][2048][1024] f16
    const float*    __restrict__ bsum,    // [2][2048]
    _Float16* __restrict__ hst0, float* __restrict__ cst0,
    _Float16* __restrict__ hst1, float* __restrict__ cst1,
    _Float16* __restrict__ h1buf,         // [2][B][H] f16 double buffer
    unsigned* __restrict__ bar,           // [0]=count [1]=generation
    float*    __restrict__ out)           // (B,H) f32
{
    __shared__ _Float16 ldsA[16 * AROW];  // 16 batch rows x K=1024 (+pad) ~33KB

    const int bid   = blockIdx.x;
    const int layer = bid >> 5;           // 0 or 1
    const int lb    = bid & 31;
    const int rb    = lb >> 2;            // 0..7  : batch-row block (16 rows)
    const int cb    = lb & 3;             // 0..3  : h-column block (128 cols)
    const int rbase = rb * 16;
    const int tid   = threadIdx.x;
    const int wv    = tid >> 5;           // wave 0..7
    const int lane  = tid & 31;
    const int laneN = lane & 15;          // C/D column within tile; A row
    const int kh    = lane >> 4;          // K-half selector for A/B fragments
    const int hcol  = cb * 128 + wv * 16; // this wave's h-column base

    const _Float16* wp  = wpack + (size_t)layer * G4H * Kn;
    const float*    bs  = bsum + layer * G4H;
    _Float16*       hst = layer ? hst1 : hst0;
    float*          cst = layer ? cst1 : cst0;

    // per-lane bias value for each gate's column
    float bv[4];
#pragma unroll
    for (int gi = 0; gi < 4; ++gi) bv[gi] = bs[gi * Hn + hcol + laneN];

    for (int s = 0; s <= Tn; ++s) {
        const bool active = (layer == 0) ? (s < Tn) : (s >= 1);
        const int  t      = (layer == 0) ? s : (s - 1);

        if (active) {
            // ---- stage A operand: rows rbase..rbase+15, K = [x(512) | h(512)] ----
            if (layer == 0) {
                const float* xs = x + ((size_t)t * Bn + rbase) * 512;
                for (int i = tid; i < 16 * 512; i += 256) {
                    int r = i >> 9, k = i & 511;
                    ldsA[r * AROW + k] = (_Float16)xs[(size_t)r * 512 + k];
                }
            } else {
                const _Float16* xs = h1buf + (size_t)(t & 1) * Bn * Hn + (size_t)rbase * Hn;
                for (int i = tid; i < 16 * 512; i += 256) {
                    int r = i >> 9, k = i & 511;
                    ldsA[r * AROW + k] = xs[(size_t)r * Hn + k];
                }
            }
            {
                const _Float16* hs = hst + (size_t)rbase * Hn;
                for (int i = tid; i < 16 * 512; i += 256) {
                    int r = i >> 9, k = i & 511;
                    ldsA[r * AROW + 512 + k] = hs[(size_t)r * Hn + k];
                }
            }
            __syncthreads();

            // ---- 4 gate tiles, accumulate over K=1024 in steps of 32 ----
            fvec8 acc[4];
#pragma unroll
            for (int gi = 0; gi < 4; ++gi) {
                fvec8 a;
#pragma unroll
                for (int r = 0; r < 8; ++r) a[r] = bv[gi];
                acc[gi] = a;
            }

            const _Float16* ldsRow = ldsA + (size_t)laneN * AROW;
#pragma unroll 4
            for (int k0 = 0; k0 < Kn; k0 += 32) {
                // A fragment: lane laneN holds row laneN; K contiguous, k = 16*kh + s (same order as B)
                hvec8 alo = *(const hvec8*)(ldsRow + k0 + kh * 16);       // K 0-7 / 16-23
                hvec8 ahi = *(const hvec8*)(ldsRow + k0 + kh * 16 + 8);   // K 8-15 / 24-31
                hvec16 afrag = __builtin_shufflevector(alo, ahi,
                    0, 1, 2, 3, 4, 5, 6, 7, 8, 9, 10, 11, 12, 13, 14, 15);
#pragma unroll
                for (int gi = 0; gi < 4; ++gi) {
                    // B fragment: lane holds column (gate col), 16 contiguous K values
                    const _Float16* bp = wp + (size_t)(gi * Hn + hcol + laneN) * Kn
                                            + k0 + kh * 16;
                    hvec16 bfrag = *(const hvec16*)bp;
                    acc[gi] = __builtin_amdgcn_wmma_f32_16x16x32_f16(
                        false, afrag, false, bfrag, (short)0, acc[gi], false, false);
                }
            }

            // ---- LSTM cell update in-registers (i,f,g,o in acc[0..3]) ----
#pragma unroll
            for (int r = 0; r < 8; ++r) {
                int b = rbase + r + kh * 8;       // C-layout: lanes 16-31 hold M = r+8
                int h = hcol + laneN;
                size_t idx = (size_t)b * Hn + h;
                float iv = fast_sigmoid(acc[0][r]);
                float fv = fast_sigmoid(acc[1][r]);
                float gv = fast_tanh(acc[2][r]);
                float ov = fast_sigmoid(acc[3][r]);
                float c  = fv * cst[idx] + iv * gv;
                float hn = ov * fast_tanh(c);
                cst[idx] = c;
                hst[idx] = (_Float16)hn;
                if (layer == 0) {
                    h1buf[(size_t)(t & 1) * Bn * Hn + idx] = (_Float16)hn;
                } else if (t == Tn - 1) {
                    out[idx] = hn;
                }
            }
        }

        // ---- grid-wide step barrier (all 64 blocks, every iteration) ----
        __threadfence();      // flush this thread's global stores to device scope
        __syncthreads();
        if (tid == 0) {
            unsigned g = __hip_atomic_load(&bar[1], __ATOMIC_RELAXED, __HIP_MEMORY_SCOPE_AGENT);
            unsigned arr = __hip_atomic_fetch_add(&bar[0], 1u, __ATOMIC_ACQ_REL,
                                                  __HIP_MEMORY_SCOPE_AGENT);
            if (arr == NBLK - 1) {
                __hip_atomic_store(&bar[0], 0u, __ATOMIC_RELAXED, __HIP_MEMORY_SCOPE_AGENT);
                __hip_atomic_fetch_add(&bar[1], 1u, __ATOMIC_RELEASE, __HIP_MEMORY_SCOPE_AGENT);
            } else {
                while (__hip_atomic_load(&bar[1], __ATOMIC_ACQUIRE,
                                         __HIP_MEMORY_SCOPE_AGENT) == g) {
                    __builtin_amdgcn_s_sleep(1);
                }
            }
        }
        __syncthreads();
    }
}

// ---------------------------------------------------------------------------
extern "C" void kernel_launch(void* const* d_in, const int* in_sizes, int n_in,
                              void* d_out, int out_size, void* d_ws, size_t ws_size,
                              hipStream_t stream)
{
    const float* x    = (const float*)d_in[0];
    const float* Wih0 = (const float*)d_in[1];
    const float* Whh0 = (const float*)d_in[2];
    const float* bih0 = (const float*)d_in[3];
    const float* bhh0 = (const float*)d_in[4];
    const float* Wih1 = (const float*)d_in[5];
    const float* Whh1 = (const float*)d_in[6];
    const float* bih1 = (const float*)d_in[7];
    const float* bhh1 = (const float*)d_in[8];

    char* ws = (char*)d_ws;
    size_t off = 0;
    auto take = [&](size_t bytes) -> char* {
        char* p = ws + off;
        off = (off + bytes + 255) & ~(size_t)255;
        return p;
    };

    _Float16* wpack = (_Float16*)take((size_t)2 * G4H * Kn * sizeof(_Float16)); // 8 MB
    float*    bsum  = (float*)take((size_t)2 * G4H * sizeof(float));

    char* zstart = ws + off;   // everything below gets zeroed each launch
    _Float16* hst0  = (_Float16*)take((size_t)Bn * Hn * sizeof(_Float16));
    float*    cst0  = (float*)take((size_t)Bn * Hn * sizeof(float));
    _Float16* hst1  = (_Float16*)take((size_t)Bn * Hn * sizeof(_Float16));
    float*    cst1  = (float*)take((size_t)Bn * Hn * sizeof(float));
    _Float16* h1buf = (_Float16*)take((size_t)2 * Bn * Hn * sizeof(_Float16));
    unsigned* bar   = (unsigned*)take(256);
    char* zend = ws + off;
    size_t zwords = (size_t)(zend - zstart) / 4;

    // 1) pack weights / biases to f16 K-contiguous layout
    {
        unsigned total = 2u * G4H * Kn;
        pack_weights<<<(total + 255) / 256, 256, 0, stream>>>(
            Wih0, Whh0, bih0, bhh0, Wih1, Whh1, bih1, bhh1, wpack, bsum);
    }
    // 2) zero h/c state, h1 staging, and barrier counters (deterministic replay)
    zero_u32<<<(unsigned)((zwords + 255) / 256), 256, 0, stream>>>((unsigned*)zstart, zwords);

    // 3) persistent fused 2-layer LSTM
    lstm_persistent<<<NBLK, 256, 0, stream>>>(
        x, wpack, bsum, hst0, cst0, hst1, cst1, h1buf, bar, (float*)d_out);
}
